// BrahmanAttention_11201274708085
// MI455X (gfx1250) — matrix operand order repaired and checked by reading, not compiled, not run on hardware
//
#include <hip/hip_runtime.h>
#include <hip/hip_bf16.h>

// ---------------------------------------------------------------------------
// BrahmanAttention for MI455X (gfx1250): bf16 WMMA end-to-end.
//   B=4, L=1024, D=1024, H=16, HD=64, NUM_CASES=8, SCALE=8.0
// Round 2: 32x64 GEMM wave-tiles (2x WMMA density), block-cooperative
// flash attention with async global->LDS staging (ASYNCcnt) + double buffer.
// ---------------------------------------------------------------------------

typedef __attribute__((ext_vector_type(16))) __bf16 v16bf;
typedef __attribute__((ext_vector_type(8)))  float  v8f;

union BF16x16 { v16bf v; uint4 u[2]; };

__device__ __forceinline__ __bf16 f2bf(float f) {
  union { float f; unsigned u; } in; in.f = f;
  unsigned u = in.u;
  u += 0x7FFFu + ((u >> 16) & 1u);            // round-to-nearest-even
  union { unsigned short s; __bf16 b; } out;
  out.s = (unsigned short)(u >> 16);
  return out.b;
}

// Load a 16-element bf16 fragment from two contiguous 16-byte chunks.
__device__ __forceinline__ v16bf load_frag(const __bf16* p0, const __bf16* p1) {
  BF16x16 r;
  r.u[0] = *(const uint4*)p0;
  r.u[1] = *(const uint4*)p1;
  return r.v;
}

__device__ __forceinline__ v8f wmma_bf16(v16bf a, v16bf b, v8f c) {
  return __builtin_amdgcn_wmma_f32_16x16x32_bf16(
      /*neg_a=*/false, a, /*neg_b=*/false, b,
      /*c_mod=*/(short)0, c, /*reuse_a=*/false, /*reuse_b=*/false);
}

// CDNA5 async global->LDS copy, 16 bytes per lane, tracked by ASYNCcnt.
__device__ __forceinline__ void async_copy16(unsigned lds_dst, const void* gsrc) {
  asm volatile("global_load_async_to_lds_b128 %0, %1, off"
               :: "v"(lds_dst), "v"((unsigned long long)(uintptr_t)gsrc)
               : "memory");
}
__device__ __forceinline__ void wait_async0() {
  asm volatile("s_wait_asynccnt 0x0" ::: "memory");
}
__device__ __forceinline__ unsigned lds_addr_of(const void* p) {
  return (unsigned)(uintptr_t)p;   // LDS aperture: addr[31:0] = LDS byte offset
}

// ---------------------------------------------------------------------------
// Kernel 1a: fp32 -> bf16 straight convert (x)
// ---------------------------------------------------------------------------
__global__ __launch_bounds__(256) void cvt_x_kernel(const float* __restrict__ in,
                                                    __bf16* __restrict__ out, int n) {
  int i = blockIdx.x * 256 + threadIdx.x;
  if (i < n) out[i] = f2bf(in[i]);
}

// ---------------------------------------------------------------------------
// Kernel 1b: fp32 [K=1024][N=1024] -> bf16 transposed [N][K]
// ---------------------------------------------------------------------------
__global__ __launch_bounds__(256) void cvt_wt_kernel(const float* __restrict__ in,
                                                     __bf16* __restrict__ out) {
  int i = blockIdx.x * 256 + threadIdx.x;   // 0 .. 1M-1
  int n = i & 1023, k = i >> 10;
  out[(size_t)n * 1024 + k] = f2bf(in[(size_t)k * 1024 + n]);
}

// ---------------------------------------------------------------------------
// Kernel 2: QKV projections. One wave computes a 32x64 tile of one of Q/K/V.
//   A = xb (bf16, row-major [4096][1024]); B = Wt (bf16, [N][K]).
//   Q,K stored head-major bf16 [b][h][L][64]; V stored transposed [b][h][64][L].
// ---------------------------------------------------------------------------
__global__ __launch_bounds__(256) void qkv_kernel(
    const __bf16* __restrict__ xb,
    const __bf16* __restrict__ Wqt, const __bf16* __restrict__ Wkt,
    const __bf16* __restrict__ Wvt,
    const float* __restrict__ bq, const float* __restrict__ bk,
    const float* __restrict__ bv,
    __bf16* __restrict__ Qh, __bf16* __restrict__ Kh, __bf16* __restrict__ Vt) {
  const int lane = threadIdx.x & 31;
  const int wave = blockIdx.x * 8 + (threadIdx.x >> 5);
  const int sel  = wave >> 11;          // 0=Q 1=K 2=V (2048 tiles each)
  const int t    = wave & 2047;         // 128 x 16 tiles of 32x64
  const int i0   = (t >> 4) * 32;
  const int n0   = (t & 15) * 64;
  const int half = lane >> 4, lr = lane & 15;

  const __bf16* Wt   = (sel == 0) ? Wqt : ((sel == 1) ? Wkt : Wvt);
  const float*  bias = (sel == 0) ? bq  : ((sel == 1) ? bk  : bv);

  const __bf16* arow0 = xb + (size_t)(i0 + lr) * 1024;
  const __bf16* arow1 = xb + (size_t)(i0 + 16 + lr) * 1024;
  v8f acc[2][4] = {};

  for (int k0 = 0; k0 < 1024; k0 += 32) {
    __builtin_prefetch(arow0 + k0 + 32, 0, 3);
    __builtin_prefetch(arow1 + k0 + 32, 0, 3);
    v16bf a0 = load_frag(arow0 + k0 + half * 8, arow0 + k0 + 16 + half * 8);
    v16bf a1 = load_frag(arow1 + k0 + half * 8, arow1 + k0 + 16 + half * 8);
#pragma unroll
    for (int d = 0; d < 4; ++d) {
      const __bf16* brow = Wt + (size_t)(n0 + d * 16 + lr) * 1024 + k0;
      v16bf bfrg = load_frag(brow + half * 8, brow + 16 + half * 8);
      acc[0][d] = wmma_bf16(a0, bfrg, acc[0][d]);
      acc[1][d] = wmma_bf16(a1, bfrg, acc[1][d]);
    }
  }

#pragma unroll
  for (int g = 0; g < 2; ++g) {
#pragma unroll
    for (int d = 0; d < 4; ++d) {
      int   ng = n0 + d * 16 + lr;
      float bb = bias[ng];
      int   h = ng >> 6, hd = ng & 63;
#pragma unroll
      for (int r = 0; r < 8; ++r) {
        int row = i0 + g * 16 + r + 8 * half;   // 0..4095
        int b = row >> 10, l = row & 1023;
        __bf16 o = f2bf(acc[g][d][r] + bb);
        if (sel == 2)
          Vt[((size_t)((b * 16 + h) * 64 + hd)) * 1024 + l] = o;
        else if (sel == 1)
          Kh[((size_t)((b * 16 + h) * 1024 + l)) * 64 + hd] = o;
        else
          Qh[((size_t)((b * 16 + h) * 1024 + l)) * 64 + hd] = o;
      }
    }
  }
}

// ---------------------------------------------------------------------------
// Kernel 3: flash attention. Block = one (b,h) + 8 i-tiles (one per wave).
// K/V tiles async-staged to LDS once per block (double-buffered, ASYNCcnt).
// ---------------------------------------------------------------------------
__global__ __launch_bounds__(256) void attn_kernel(
    const __bf16* __restrict__ Qh, const __bf16* __restrict__ Kh,
    const __bf16* __restrict__ Vt,
    const int* __restrict__ case_ids, const int* __restrict__ verb_mask,
    const float* __restrict__ case_bias, const float* __restrict__ verb_bias,
    __bf16* __restrict__ xo) {
  __shared__ float s_cb[64];                               // this head's 8x8 bias
  __shared__ __align__(16) __bf16 s_k[2][32 * 64];         // K tile: [j][hd]
  __shared__ __align__(16) __bf16 s_v[2][64 * 32];         // V tile: [hd][j]
  __shared__ __align__(16) __bf16 s_p[8][16 * 32];         // per-wave P tile

  const int lane  = threadIdx.x & 31;
  const int wslot = threadIdx.x >> 5;
  const int bh  = blockIdx.x >> 3;                         // 0..63
  const int grp = blockIdx.x & 7;
  const int b = bh >> 4, h = bh & 15;
  const int i0 = (grp * 8 + wslot) * 16;
  const int half = lane >> 4, lr = lane & 15;
  const float inv_scale = 0.125f;                          // 1/sqrt(64)

  const __bf16* Qbase = Qh + (size_t)(b * 16 + h) * 1024 * 64;
  const __bf16* Kbase = Kh + (size_t)(b * 16 + h) * 1024 * 64;
  const __bf16* Vbase = Vt + (size_t)(b * 16 + h) * 64 * 1024;
  const int* cid = case_ids + b * 1024;
  const int* vm  = verb_mask + b * 1024;
  const float vb_h = verb_bias[h];
  __bf16* pl = &s_p[wslot][0];

  if (threadIdx.x < 64) s_cb[threadIdx.x] = case_bias[h * 64 + threadIdx.x];

  // cooperative async staging of one 32-wide j-chunk: K 32x64, V 64x32
  const int kr = threadIdx.x >> 3, kc = threadIdx.x & 7;   // 32 rows x 8 chunks
  const int vr = threadIdx.x >> 2, vc = threadIdx.x & 3;   // 64 rows x 4 chunks
  const unsigned kdst[2] = { lds_addr_of(&s_k[0][0]) + (unsigned)(kr * 128 + kc * 16),
                             lds_addr_of(&s_k[1][0]) + (unsigned)(kr * 128 + kc * 16) };
  const unsigned vdst[2] = { lds_addr_of(&s_v[0][0]) + (unsigned)(vr * 64 + vc * 16),
                             lds_addr_of(&s_v[1][0]) + (unsigned)(vr * 64 + vc * 16) };
#define STAGE_KV(bufi, jbase)                                                   \
  do {                                                                          \
    async_copy16(kdst[bufi], Kbase + (size_t)((jbase) + kr) * 64 + kc * 8);     \
    async_copy16(vdst[bufi], Vbase + (size_t)vr * 1024 + (jbase) + vc * 8);     \
  } while (0)

  // Resident Q A-fragments (K = 0..31 and 32..63)
  const __bf16* qrow = Qbase + (size_t)(i0 + lr) * 64;
  v16bf qa0 = load_frag(qrow +      half * 8, qrow + 16 + half * 8);
  v16bf qa1 = load_frag(qrow + 32 + half * 8, qrow + 48 + half * 8);

  int crow[8];
#pragma unroll
  for (int r = 0; r < 8; ++r) crow[r] = cid[i0 + r + 8 * half];

  float rmax[8], rsum[8];
#pragma unroll
  for (int r = 0; r < 8; ++r) { rmax[r] = -3.0e38f; rsum[r] = 0.f; }
  v8f acc[4] = {};

  STAGE_KV(0, 0);
  int buf = 0;
  for (int jj = 0; jj < 1024; jj += 32) {
    wait_async0();          // our async copies into s_k/s_v[buf] are done
    __syncthreads();        // everyone's copies are done -> tile[buf] ready
    if (jj + 32 < 1024) STAGE_KV(buf ^ 1, jj + 32);

    const __bf16* kb = &s_k[buf][0];
    const __bf16* vb = &s_v[buf][0];

    float s0[8], s1[8];
    // -- two 16-wide score tiles: S = Q x K^T / scale + biases
#pragma unroll
    for (int t = 0; t < 2; ++t) {
      const __bf16* krow = kb + (t * 16 + lr) * 64;        // staged [j][hd]
      v16bf bk0 = load_frag(krow +      half * 8, krow + 16 + half * 8);
      v16bf bk1 = load_frag(krow + 32 + half * 8, krow + 48 + half * 8);
      v8f c = {};
      c = wmma_bf16(qa0, bk0, c);
      c = wmma_bf16(qa1, bk1, c);
      const int j    = jj + t * 16 + lr;
      const int ccol = cid[j];
      const float vbj = vb_h * (float)vm[j];
      float* s = t ? s1 : s0;
#pragma unroll
      for (int r = 0; r < 8; ++r)
        s[r] = c[r] * inv_scale + s_cb[crow[r] * 8 + ccol] + vbj;
    }

    // -- online softmax: row max/sum via 16-lane butterflies
    float nmax[8], ascale[8];
#pragma unroll
    for (int r = 0; r < 8; ++r) {
      float m = fmaxf(s0[r], s1[r]);
      m = fmaxf(m, __shfl_xor(m, 1, 16));
      m = fmaxf(m, __shfl_xor(m, 2, 16));
      m = fmaxf(m, __shfl_xor(m, 4, 16));
      m = fmaxf(m, __shfl_xor(m, 8, 16));
      nmax[r]   = fmaxf(rmax[r], m);
      ascale[r] = __expf(rmax[r] - nmax[r]);
      rmax[r]   = nmax[r];
    }
#pragma unroll
    for (int r = 0; r < 8; ++r) {
      float p0 = __expf(s0[r] - nmax[r]);
      float p1 = __expf(s1[r] - nmax[r]);
      float ps = p0 + p1;
      ps += __shfl_xor(ps, 1, 16);
      ps += __shfl_xor(ps, 2, 16);
      ps += __shfl_xor(ps, 4, 16);
      ps += __shfl_xor(ps, 8, 16);
      rsum[r] = rsum[r] * ascale[r] + ps;
      const int m = r + 8 * half;
      pl[m * 32 +      lr] = f2bf(p0);
      pl[m * 32 + 16 + lr] = f2bf(p1);
#pragma unroll
      for (int d = 0; d < 4; ++d) acc[d][r] *= ascale[r];
    }

    // -- read P back in A-fragment layout (DS ops are in-order per wave)
    v16bf pa = load_frag(pl + lr * 32 + half * 8, pl + lr * 32 + 16 + half * 8);

    // -- O += P x V  (staged V: [hd][j], contiguous along j)
#pragma unroll
    for (int d = 0; d < 4; ++d) {
      const __bf16* vrow = vb + (d * 16 + lr) * 32;
      acc[d] = wmma_bf16(pa, load_frag(vrow + half * 8, vrow + 16 + half * 8), acc[d]);
    }

    __syncthreads();        // all reads of tile[buf] done before it is restaged
    buf ^= 1;
  }

  // -- normalize + store bf16 attention output [4096][1024]
#pragma unroll
  for (int r = 0; r < 8; ++r) {
    float inv = 1.0f / rsum[r];
    int row = b * 1024 + i0 + r + 8 * half;
#pragma unroll
    for (int d = 0; d < 4; ++d)
      xo[(size_t)row * 1024 + h * 64 + d * 16 + lr] = f2bf(acc[d][r] * inv);
  }
#undef STAGE_KV
}

// ---------------------------------------------------------------------------
// Kernel 4: output projection: out = xo x Wo + bo (fp32), 32x64 wave-tiles
// ---------------------------------------------------------------------------
__global__ __launch_bounds__(256) void out_kernel(
    const __bf16* __restrict__ xo, const __bf16* __restrict__ Wot,
    const float* __restrict__ bo, float* __restrict__ out) {
  const int lane = threadIdx.x & 31;
  const int wave = blockIdx.x * 8 + (threadIdx.x >> 5);   // 0..2047
  const int i0 = (wave >> 4) * 32;
  const int n0 = (wave & 15) * 64;
  const int half = lane >> 4, lr = lane & 15;

  const __bf16* arow0 = xo + (size_t)(i0 + lr) * 1024;
  const __bf16* arow1 = xo + (size_t)(i0 + 16 + lr) * 1024;
  v8f acc[2][4] = {};

  for (int k0 = 0; k0 < 1024; k0 += 32) {
    __builtin_prefetch(arow0 + k0 + 32, 0, 3);
    __builtin_prefetch(arow1 + k0 + 32, 0, 3);
    v16bf a0 = load_frag(arow0 + k0 + half * 8, arow0 + k0 + 16 + half * 8);
    v16bf a1 = load_frag(arow1 + k0 + half * 8, arow1 + k0 + 16 + half * 8);
#pragma unroll
    for (int d = 0; d < 4; ++d) {
      const __bf16* brow = Wot + (size_t)(n0 + d * 16 + lr) * 1024 + k0;
      v16bf bfrg = load_frag(brow + half * 8, brow + 16 + half * 8);
      acc[0][d] = wmma_bf16(a0, bfrg, acc[0][d]);
      acc[1][d] = wmma_bf16(a1, bfrg, acc[1][d]);
    }
  }

#pragma unroll
  for (int g = 0; g < 2; ++g) {
#pragma unroll
    for (int d = 0; d < 4; ++d) {
      int   ng = n0 + d * 16 + lr;
      float bb = bo[ng];
#pragma unroll
      for (int r = 0; r < 8; ++r) {
        int row = i0 + g * 16 + r + 8 * half;
        out[(size_t)row * 1024 + ng] = acc[g][d][r] + bb;
      }
    }
  }
}

// ---------------------------------------------------------------------------
// Host launcher
// ---------------------------------------------------------------------------
extern "C" void kernel_launch(void* const* d_in, const int* in_sizes, int n_in,
                              void* d_out, int out_size, void* d_ws, size_t ws_size,
                              hipStream_t stream) {
  const float* x         = (const float*)d_in[0];
  const int*   case_ids  = (const int*)d_in[1];
  const int*   verb_mask = (const int*)d_in[2];
  const float* Wq        = (const float*)d_in[3];
  const float* bq        = (const float*)d_in[4];
  const float* Wk        = (const float*)d_in[5];
  const float* bk        = (const float*)d_in[6];
  const float* Wv        = (const float*)d_in[7];
  const float* bv        = (const float*)d_in[8];
  const float* Wo        = (const float*)d_in[9];
  const float* bo        = (const float*)d_in[10];
  const float* case_bias = (const float*)d_in[11];
  const float* verb_bias = (const float*)d_in[12];
  float* out = (float*)d_out;

  char* ws = (char*)d_ws;
  __bf16* xb  = (__bf16*)(ws + (size_t)0);
  __bf16* Wqt = (__bf16*)(ws + ((size_t)8  << 20));
  __bf16* Wkt = (__bf16*)(ws + ((size_t)10 << 20));
  __bf16* Wvt = (__bf16*)(ws + ((size_t)12 << 20));
  __bf16* Wot = (__bf16*)(ws + ((size_t)14 << 20));
  __bf16* Qh  = (__bf16*)(ws + ((size_t)16 << 20));
  __bf16* Kh  = (__bf16*)(ws + ((size_t)24 << 20));
  __bf16* Vt  = (__bf16*)(ws + ((size_t)32 << 20));
  __bf16* xo  = (__bf16*)(ws + ((size_t)40 << 20));

  // 1) precision conversion (+ weight transposes)
  cvt_x_kernel<<<16384, 256, 0, stream>>>(x, xb, 4 * 1024 * 1024);
  cvt_wt_kernel<<<4096, 256, 0, stream>>>(Wq, Wqt);
  cvt_wt_kernel<<<4096, 256, 0, stream>>>(Wk, Wkt);
  cvt_wt_kernel<<<4096, 256, 0, stream>>>(Wv, Wvt);
  cvt_wt_kernel<<<4096, 256, 0, stream>>>(Wo, Wot);

  // 2) Q/K/V projections: 3 * 2048 wave-tiles (32x64), 8 waves/block
  qkv_kernel<<<768, 256, 0, stream>>>(xb, Wqt, Wkt, Wvt, bq, bk, bv, Qh, Kh, Vt);

  // 3) flash attention: 64 (b,h) blocks x 8 i-tile groups
  attn_kernel<<<512, 256, 0, stream>>>(Qh, Kh, Vt, case_ids, verb_mask,
                                       case_bias, verb_bias, xo);

  // 4) output projection: 2048 wave-tiles (32x64)
  out_kernel<<<256, 256, 0, stream>>>(xo, Wot, bo, out);
}
